// CrossAttentionFusion_85701777424912
// MI455X (gfx1250) — compile-verified, matrix operand order checked by reading, not run on hardware
//
#include <hip/hip_runtime.h>
#include <cstdint>
#include <cstddef>

// ---------------------------------------------------------------------------
// Types for CDNA5 WMMA (wave32): v_wmma_f32_16x16x32_bf16
// ---------------------------------------------------------------------------
typedef __attribute__((ext_vector_type(16))) __bf16 v16bf;
typedef __attribute__((ext_vector_type(8)))  float  v8f;

static constexpr int kB  = 16384;   // batch rows
static constexpr int kE  = 768;     // embed dim
static constexpr int kH4 = 3072;    // 4*E FFN hidden
static constexpr int k2E = 1536;    // concat dim
static constexpr float kEps = 1e-5f;

__device__ __forceinline__ unsigned short f32_to_bf16(float f) {
  union { float f; unsigned u; } v; v.f = f;
  unsigned u = v.u;
  if ((u & 0x7fffffffu) > 0x7f800000u) return (unsigned short)((u >> 16) | 0x40);
  unsigned r = 0x7fffu + ((u >> 16) & 1u);
  return (unsigned short)((u + r) >> 16);
}

__device__ __forceinline__ float gelu_exact(float x) {
  return 0.5f * x * (1.0f + erff(x * 0.70710678118654752f));
}

// CDNA5 async copy: global -> LDS, tracked by ASYNCcnt. GVS addressing:
//   mem = SADDR(base) + VADDR(voff) + INST_OFFSET ; lds = VDST + INST_OFFSET
__device__ __forceinline__ void async_b128(unsigned lds_addr, unsigned voff,
                                           const void* sbase) {
  asm volatile("global_load_async_to_lds_b128 %0, %1, %2 offset:0"
               :: "v"(lds_addr), "v"(voff), "s"(sbase) : "memory");
}
__device__ __forceinline__ void async_b128_o16(unsigned lds_addr, unsigned voff,
                                               const void* sbase) {
  asm volatile("global_load_async_to_lds_b128 %0, %1, %2 offset:16"
               :: "v"(lds_addr), "v"(voff), "s"(sbase) : "memory");
}
__device__ __forceinline__ void wait_async0() {
  asm volatile("s_wait_asynccnt 0x0" ::: "memory");
}

// ---------------------------------------------------------------------------
// f32 -> bf16 bulk convert (vectorized x4)
// ---------------------------------------------------------------------------
__global__ void cvt_f32_to_bf16(const float* __restrict__ in,
                                unsigned short* __restrict__ out, int n4) {
  int i = blockIdx.x * blockDim.x + threadIdx.x;
  if (i >= n4) return;
  float4 x = ((const float4*)in)[i];
  ushort4 y;
  y.x = f32_to_bf16(x.x); y.y = f32_to_bf16(x.y);
  y.z = f32_to_bf16(x.z); y.w = f32_to_bf16(x.w);
  ((ushort4*)out)[i] = y;
}

// ---------------------------------------------------------------------------
// Attention weight fold: out[n,k] = sum_j Wo[n,j] * Wv[j,k]   (768x768x768)
// (softmax over a single key == 1, so mha1 collapses to one GEMM)
// ---------------------------------------------------------------------------
__global__ __launch_bounds__(256)
void fold_w(const float* __restrict__ Wo, const float* __restrict__ Wv,
            unsigned short* __restrict__ outBf) {
  __shared__ float to_[16][17], tv_[16][17];
  const int tx = threadIdx.x, ty = threadIdx.y;
  const int n = blockIdx.y * 16 + ty;
  const int k = blockIdx.x * 16 + tx;
  float s = 0.f;
  for (int j0 = 0; j0 < kE; j0 += 16) {
    to_[ty][tx] = Wo[(size_t)n * kE + j0 + tx];
    tv_[ty][tx] = Wv[(size_t)(j0 + ty) * kE + k];
    __syncthreads();
#pragma unroll
    for (int j = 0; j < 16; ++j) s += to_[ty][j] * tv_[j][tx];
    __syncthreads();
  }
  outBf[(size_t)n * kE + k] = f32_to_bf16(s);
}

__global__ void fold_bias(const float* __restrict__ Wo,
                          const float* __restrict__ bv,
                          const float* __restrict__ bo,
                          float* __restrict__ out) {
  const int n = blockIdx.x * blockDim.x + threadIdx.x;
  if (n >= kE) return;
  float s = bo[n];
  for (int j = 0; j < kE; ++j) s += Wo[(size_t)n * kE + j] * bv[j];
  out[n] = s;
}

// ---------------------------------------------------------------------------
// Main bf16 WMMA GEMM:  C[M,N] = A[M,K] @ W[N,K]^T  (+bias +res, opt. gelu)
//   block tile 128x128, BK=32, 256 threads = 8 waves (4 M x 2 N),
//   wave tile 32x64 = 2x4 v_wmma_f32_16x16x32_bf16 accumulators.
//   Double-buffered LDS filled with global_load_async_to_lds_b128 (ASYNCcnt).
// ---------------------------------------------------------------------------
#define BM 128
#define BN 128
#define BK 32
#define LP 40   // padded LDS row pitch in bf16 elements (80 bytes)

union FragU { v16bf v; uint4 q[2]; };

__global__ __launch_bounds__(256)
void gemm_bf16(const unsigned short* __restrict__ A, int lda,
               const unsigned short* __restrict__ W,   // [N,K], row-major
               const float* __restrict__ bias,          // [N] or null
               const float* __restrict__ res,           // [M,N] f32 or null
               float* __restrict__ outF,                // [M,N] f32 or null
               unsigned short* __restrict__ outB, int ldob,  // bf16 out or null
               int N, int K, int geluFlag) {
  __shared__ unsigned short As[2][BM * LP];
  __shared__ unsigned short Bs[2][BN * LP];

  const int tid  = threadIdx.x;
  const int lane = tid & 31;
  const int wave = tid >> 5;
  const int wm = wave & 3;          // wave M index (0..3) -> 32 rows each
  const int wn = wave >> 2;         // wave N index (0..1) -> 64 cols each
  const int bm = blockIdx.y * BM;
  const int bn = blockIdx.x * BN;

  // async staging: each thread moves 32B of A and 32B of W per tile
  const int lr = tid >> 1;          // tile row 0..127
  const int lc = (tid & 1) << 4;    // element offset 0 or 16
  const unsigned voffA0 = (unsigned)(((size_t)(bm + lr) * lda + lc) * 2);
  const unsigned voffW0 = (unsigned)(((size_t)(bn + lr) * K   + lc) * 2);
  const unsigned sOffB  = (unsigned)((lr * LP + lc) * 2);   // bytes into buffer
  const unsigned ldsA0  = (unsigned)(size_t)&As[0][0];
  const unsigned ldsB0  = (unsigned)(size_t)&Bs[0][0];
  const unsigned bufBytes = BM * LP * 2;

  // stage tile 0 (async DMA into LDS)
  async_b128    (ldsA0 + sOffB, voffA0, A);
  async_b128_o16(ldsA0 + sOffB, voffA0, A);
  async_b128    (ldsB0 + sOffB, voffW0, W);
  async_b128_o16(ldsB0 + sOffB, voffW0, W);
  wait_async0();
  __syncthreads();

  v8f acc[2][4];
  {
    v8f z;
#pragma unroll
    for (int e = 0; e < 8; ++e) z[e] = 0.f;
#pragma unroll
    for (int i = 0; i < 2; ++i)
#pragma unroll
      for (int j = 0; j < 4; ++j) acc[i][j] = z;
  }

  // fragment addressing per ISA bf16 A/B layout:
  //  lane<16 holds K {0..7,16..23}, lane>=16 holds K {8..15,24..31}
  const int laneRow  = lane & 15;
  const int laneHalf = lane >> 4;
  const int aBase0 = (wm * 32 + laneRow) * LP + laneHalf * 8;
  const int bBase0 = (wn * 64 + laneRow) * LP + laneHalf * 8;

  const int KT = K / BK;
  for (int kt = 0; kt < KT; ++kt) {
    const int cur = kt & 1;
    const bool more = (kt + 1 < KT);
    if (more) {
      // kick off async DMA for the next K-tile into the other buffer
      const unsigned kb = (unsigned)((kt + 1) * BK * 2);
      const unsigned la = ldsA0 + (unsigned)(cur ^ 1) * bufBytes + sOffB;
      const unsigned lb = ldsB0 + (unsigned)(cur ^ 1) * bufBytes + sOffB;
      async_b128    (la, voffA0 + kb, A);
      async_b128_o16(la, voffA0 + kb, A);
      async_b128    (lb, voffW0 + kb, W);
      async_b128_o16(lb, voffW0 + kb, W);
    }

    FragU af[2], bfr[4];
#pragma unroll
    for (int fm = 0; fm < 2; ++fm) {
      const int base = aBase0 + fm * 16 * LP;
      af[fm].q[0] = *(const uint4*)&As[cur][base];
      af[fm].q[1] = *(const uint4*)&As[cur][base + 16];
    }
#pragma unroll
    for (int fn = 0; fn < 4; ++fn) {
      const int base = bBase0 + fn * 16 * LP;
      bfr[fn].q[0] = *(const uint4*)&Bs[cur][base];
      bfr[fn].q[1] = *(const uint4*)&Bs[cur][base + 16];
    }
#pragma unroll
    for (int fm = 0; fm < 2; ++fm)
#pragma unroll
      for (int fn = 0; fn < 4; ++fn)
        acc[fm][fn] = __builtin_amdgcn_wmma_f32_16x16x32_bf16(
            false, af[fm].v, false, bfr[fn].v, (short)0, acc[fm][fn],
            false, false);

    if (more) wait_async0();   // own async DMAs done; barrier covers the rest
    __syncthreads();
  }

  // epilogue: C layout VGPR v: lanes 0-15 -> M=v, lanes 16-31 -> M=v+8
#pragma unroll
  for (int fm = 0; fm < 2; ++fm) {
    const int row0 = bm + wm * 32 + fm * 16 + laneHalf * 8;
#pragma unroll
    for (int fn = 0; fn < 4; ++fn) {
      const int col = bn + wn * 64 + fn * 16 + laneRow;
      const float bval = bias ? bias[col] : 0.f;
#pragma unroll
      for (int v = 0; v < 8; ++v) {
        const int row = row0 + v;
        float val = acc[fm][fn][v] + bval;
        if (res)  val += res[(size_t)row * N + col];
        if (geluFlag) val = gelu_exact(val);
        if (outF) outF[(size_t)row * N + col] = val;
        if (outB) outB[(size_t)row * ldob + col] = f32_to_bf16(val);
      }
    }
  }
}

// ---------------------------------------------------------------------------
// LayerNorm over 768 (one row per block, 3 elements/thread), opt gelu,
// emits f32 and/or bf16.
// ---------------------------------------------------------------------------
__global__ __launch_bounds__(256)
void layernorm_768(const float* __restrict__ in,
                   const float* __restrict__ g, const float* __restrict__ b,
                   float* __restrict__ outF, unsigned short* __restrict__ outB,
                   int geluFlag) {
  __shared__ float red[256];
  const int row = blockIdx.x;
  const int tid = threadIdx.x;
  const float* x = in + (size_t)row * kE;
  float v0 = x[tid], v1 = x[tid + 256], v2 = x[tid + 512];
  red[tid] = v0 + v1 + v2;
  __syncthreads();
  for (int off = 128; off > 0; off >>= 1) {
    if (tid < off) red[tid] += red[tid + off];
    __syncthreads();
  }
  const float mean = red[0] * (1.0f / 768.0f);
  __syncthreads();
  const float d0 = v0 - mean, d1 = v1 - mean, d2 = v2 - mean;
  red[tid] = d0 * d0 + d1 * d1 + d2 * d2;
  __syncthreads();
  for (int off = 128; off > 0; off >>= 1) {
    if (tid < off) red[tid] += red[tid + off];
    __syncthreads();
  }
  const float rstd = rsqrtf(red[0] * (1.0f / 768.0f) + kEps);
  float y0 = d0 * rstd * g[tid]       + b[tid];
  float y1 = d1 * rstd * g[tid + 256] + b[tid + 256];
  float y2 = d2 * rstd * g[tid + 512] + b[tid + 512];
  if (geluFlag) { y0 = gelu_exact(y0); y1 = gelu_exact(y1); y2 = gelu_exact(y2); }
  if (outF) {
    float* o = outF + (size_t)row * kE;
    o[tid] = y0; o[tid + 256] = y1; o[tid + 512] = y2;
  }
  if (outB) {
    unsigned short* o = outB + (size_t)row * kE;
    o[tid] = f32_to_bf16(y0);
    o[tid + 256] = f32_to_bf16(y1);
    o[tid + 512] = f32_to_bf16(y2);
  }
}

// ---------------------------------------------------------------------------
// Host orchestration
// ---------------------------------------------------------------------------
extern "C" void kernel_launch(void* const* d_in, const int* in_sizes, int n_in,
                              void* d_out, int out_size, void* d_ws,
                              size_t ws_size, hipStream_t stream) {
  (void)in_sizes; (void)n_in; (void)out_size; (void)ws_size;

  const float* image    = (const float*)d_in[0];
  const float* text     = (const float*)d_in[1];
  const float* it_Wv    = (const float*)d_in[4];
  const float* it_Wo    = (const float*)d_in[5];
  const float* it_bv    = (const float*)d_in[8];
  const float* it_bo    = (const float*)d_in[9];
  const float* ti_Wv    = (const float*)d_in[12];
  const float* ti_Wo    = (const float*)d_in[13];
  const float* ti_bv    = (const float*)d_in[16];
  const float* ti_bo    = (const float*)d_in[17];
  const float* ln_img_g = (const float*)d_in[18];
  const float* ln_img_b = (const float*)d_in[19];
  const float* ln_txt_g = (const float*)d_in[20];
  const float* ln_txt_b = (const float*)d_in[21];
  const float* fp_ln_g  = (const float*)d_in[22];
  const float* fp_ln_b  = (const float*)d_in[23];
  const float* fi_W1    = (const float*)d_in[24];
  const float* fi_b1    = (const float*)d_in[25];
  const float* fi_W2    = (const float*)d_in[26];
  const float* fi_b2    = (const float*)d_in[27];
  const float* ft_W1    = (const float*)d_in[28];
  const float* ft_b1    = (const float*)d_in[29];
  const float* ft_W2    = (const float*)d_in[30];
  const float* ft_b2    = (const float*)d_in[31];
  const float* fp_W     = (const float*)d_in[32];
  const float* fp_b     = (const float*)d_in[33];

  char* w = (char*)d_ws;
  auto take = [&](size_t bytes) -> char* {
    char* p = w; w += (bytes + 255) & ~(size_t)255; return p;
  };
  unsigned short* text_bf  = (unsigned short*)take((size_t)kB * kE * 2);
  unsigned short* imgln_bf = (unsigned short*)take((size_t)kB * kE * 2);
  unsigned short* txtln_bf = (unsigned short*)take((size_t)kB * kE * 2);
  unsigned short* H_bf     = (unsigned short*)take((size_t)kB * kH4 * 2);
  unsigned short* fused_bf = (unsigned short*)take((size_t)kB * k2E * 2);
  float*          tmpF     = (float*)take((size_t)kB * kE * 4);
  float*          imgln_f  = (float*)take((size_t)kB * kE * 4);
  float*          txtln_f  = (float*)take((size_t)kB * kE * 4);
  unsigned short* WitF     = (unsigned short*)take((size_t)kE * kE * 2);
  unsigned short* WtiF     = (unsigned short*)take((size_t)kE * kE * 2);
  unsigned short* fiW1b    = (unsigned short*)take((size_t)kH4 * kE * 2);
  unsigned short* fiW2b    = (unsigned short*)take((size_t)kE * kH4 * 2);
  unsigned short* ftW1b    = (unsigned short*)take((size_t)kH4 * kE * 2);
  unsigned short* ftW2b    = (unsigned short*)take((size_t)kE * kH4 * 2);
  unsigned short* fpWb     = (unsigned short*)take((size_t)kE * k2E * 2);
  float* bit = (float*)take(kE * 4);
  float* bti = (float*)take(kE * 4);

  auto cvt = [&](const float* src, unsigned short* dst, size_t n) {
    int n4 = (int)(n / 4);
    cvt_f32_to_bf16<<<(n4 + 255) / 256, 256, 0, stream>>>(src, dst, n4);
  };
  cvt(text,  text_bf, (size_t)kB * kE);
  cvt(fi_W1, fiW1b,   (size_t)kH4 * kE);
  cvt(fi_W2, fiW2b,   (size_t)kE * kH4);
  cvt(ft_W1, ftW1b,   (size_t)kH4 * kE);
  cvt(ft_W2, ftW2b,   (size_t)kE * kH4);
  cvt(fp_W,  fpWb,    (size_t)kE * k2E);

  dim3 fb(16, 16), fg(kE / 16, kE / 16);
  fold_w<<<fg, fb, 0, stream>>>(it_Wo, it_Wv, WitF);
  fold_w<<<fg, fb, 0, stream>>>(ti_Wo, ti_Wv, WtiF);
  fold_bias<<<3, 256, 0, stream>>>(it_Wo, it_bv, it_bo, bit);
  fold_bias<<<3, 256, 0, stream>>>(ti_Wo, ti_bv, ti_bo, bti);

  dim3 blk(256);
  dim3 gE(kE / BN, kB / BM);    // N = 768
  dim3 gH(kH4 / BN, kB / BM);   // N = 3072

  // att1 (folded): tmp = text @ Wit^T + bit + image
  gemm_bf16<<<gE, blk, 0, stream>>>(text_bf, kE, WitF, bit, image,
                                    tmpF, nullptr, 0, kE, kE, 0);
  // img = LN(tmp)
  layernorm_768<<<kB, 256, 0, stream>>>(tmpF, ln_img_g, ln_img_b,
                                        imgln_f, imgln_bf, 0);
  // H = gelu(img @ fiW1^T + b1)
  gemm_bf16<<<gH, blk, 0, stream>>>(imgln_bf, kE, fiW1b, fi_b1, nullptr,
                                    nullptr, H_bf, kH4, kH4, kE, 1);
  // img_final = img + H @ fiW2^T + b2   -> fused[:, :768] (bf16)
  gemm_bf16<<<gE, blk, 0, stream>>>(H_bf, kH4, fiW2b, fi_b2, imgln_f,
                                    nullptr, fused_bf, k2E, kE, kH4, 0);
  // att2 (folded): tmp = img_final @ Wti^T + bti + text
  gemm_bf16<<<gE, blk, 0, stream>>>(fused_bf, k2E, WtiF, bti, text,
                                    tmpF, nullptr, 0, kE, kE, 0);
  // txt = LN(tmp)
  layernorm_768<<<kB, 256, 0, stream>>>(tmpF, ln_txt_g, ln_txt_b,
                                        txtln_f, txtln_bf, 0);
  // H = gelu(txt @ ftW1^T + b1)
  gemm_bf16<<<gH, blk, 0, stream>>>(txtln_bf, kE, ftW1b, ft_b1, nullptr,
                                    nullptr, H_bf, kH4, kH4, kE, 1);
  // txt_final = txt + H @ ftW2^T + b2   -> fused[:, 768:] (bf16)
  gemm_bf16<<<gE, blk, 0, stream>>>(H_bf, kH4, ftW2b, ft_b2, txtln_f,
                                    nullptr, fused_bf + kE, k2E, kE, kH4, 0);
  // fusion proj: tmp = fused @ fpW^T + fp_b  (K = 1536)
  gemm_bf16<<<gE, blk, 0, stream>>>(fused_bf, k2E, fpWb, fp_b, nullptr,
                                    tmpF, nullptr, 0, kE, k2E, 0);
  // out = gelu(LN(tmp))
  layernorm_768<<<kB, 256, 0, stream>>>(tmpF, fp_ln_g, fp_ln_b,
                                        (float*)d_out, nullptr, 1);
}
